// SelfAttention_20839181320238
// MI455X (gfx1250) — matrix operand order repaired and checked by reading, not compiled, not run on hardware
//
#include <hip/hip_runtime.h>
#include <hip/hip_bf16.h>

// ---------------------------------------------------------------------------
// Problem constants: x [B=8,V=4,L=1024,C=512] fp32, d = C/4 = 128.
// BV = 32, rows = BV*L = 32768.
// ---------------------------------------------------------------------------
#define L_SEQ   1024
#define C_DIM   512
#define D_DIM   128
#define BV_N    32
#define ROWS    (BV_N * L_SEQ)   // 32768
#define LNP     520              // padded LDS row stride (floats); 520 % 64 = 8

typedef __attribute__((ext_vector_type(16))) _Float16 v16h;
typedef __attribute__((ext_vector_type(8)))  _Float16 v8h;
typedef __attribute__((ext_vector_type(8)))  float    v8f;

#define WMMA_F16(a, b, c) \
  __builtin_amdgcn_wmma_f32_16x16x32_f16(false, (a), false, (b), (short)0, (c), false, false)

static __device__ __forceinline__ int lane_id() { return (int)(threadIdx.x & 31u); }

// A-fragment (16x32 f16, row-major source, leading dim `ld` in elements).
// Layout (ISA 7.12.2): lanes 0-15 -> row M=lane, VGPR0..3 = K 0..7, VGPR4..7 = K 16..23;
// lanes 16-31 -> row M=lane-16, K 8..15 and 24..31.  => two contiguous 16B loads/lane.
static __device__ __forceinline__ v16h load_a_frag(const _Float16* __restrict__ base,
                                                   int ld, int k0) {
  const int lane = lane_id();
  const int r    = lane & 15;
  const int koff = (lane >> 4) << 3;          // 0 or 8
  const _Float16* p = base + (size_t)r * ld + (k0 + koff);
  v8h lo = *(const v8h*)(p);
  v8h hi = *(const v8h*)(p + 16);
  v16h f;
#pragma unroll
  for (int i = 0; i < 8; ++i) { f[i] = lo[i]; f[i + 8] = hi[i]; }
  return f;
}

// Same A-fragment but converting from an f32 source (LDS softmax probs).
static __device__ __forceinline__ v16h load_a_frag_f32(const float* __restrict__ base,
                                                       int ld, int k0) {
  const int lane = lane_id();
  const int r    = lane & 15;
  const int koff = (lane >> 4) << 3;
  const float* p = base + (size_t)r * ld + (k0 + koff);
  v16h f;
#pragma unroll
  for (int i = 0; i < 8; ++i) {
    f[i]     = (_Float16)p[i];
    f[i + 8] = (_Float16)p[i + 16];
  }
  return f;
}

// B-fragment (32x16 f16). Source is the *transpose* stored row-major [N, K]
// (i.e. column-major B), leading dim `ld`. Layout: lanes 0-15 -> col N=lane,
// K=0..7,16..23; lanes 16-31 -> K=8..15,24..31.  => two 16B loads per lane.
static __device__ __forceinline__ v16h load_b_frag(const _Float16* __restrict__ base,
                                                   int ld, int k0) {
  const int lane = lane_id();
  const int n    = lane & 15;
  const int koff = (lane >> 4) << 3;          // 0 or 8
  const _Float16* p = base + (size_t)n * ld + (k0 + koff);
  v8h lo = *(const v8h*)(p);
  v8h hi = *(const v8h*)(p + 16);
  v16h f;
#pragma unroll
  for (int i = 0; i < 8; ++i) { f[i] = lo[i]; f[i + 8] = hi[i]; }
  return f;
}

// ---------------------------------------------------------------------------
// Kernel 1: LayerNorm + coalesced transposed residual.
// Block = 32 consecutive rows (8 waves x 4 rows, one wave-reduction per row).
// Normalized tile staged in padded LDS [32][LNP], then the residual
//   d_out[bv, c*L + l] = xn[bv, l, c]
// is written as contiguous float4 runs (8 threads = one 128B line per channel).
// ---------------------------------------------------------------------------
__global__ __launch_bounds__(256) void ln_kernel(const float* __restrict__ x,
                                                 const float* __restrict__ ln_w,
                                                 const float* __restrict__ ln_b,
                                                 _Float16* __restrict__ xn,
                                                 float* __restrict__ resid) {
  extern __shared__ float T[];               // [32][LNP] fp32, ~65 KB
  const int r0   = blockIdx.x << 5;          // first row of this block
  const int wid  = (int)(threadIdx.x >> 5);
  const int lane = lane_id();
  const int bv   = r0 >> 10;
  const int l0   = r0 & (L_SEQ - 1);

#pragma unroll
  for (int rr = 0; rr < 4; ++rr) {
    const int ll  = wid * 4 + rr;            // local row 0..31
    const int row = r0 + ll;
    const float* xr = x + (size_t)row * C_DIM;
    float vals[16];
    float s = 0.f, ss = 0.f;
#pragma unroll
    for (int i = 0; i < 16; ++i) {
      float v = xr[lane + 32 * i];
      vals[i] = v;
      s += v;
      ss += v * v;
    }
#pragma unroll
    for (int off = 16; off; off >>= 1) {
      s  += __shfl_xor(s,  off, 32);
      ss += __shfl_xor(ss, off, 32);
    }
    const float mu  = s * (1.0f / (float)C_DIM);
    const float var = ss * (1.0f / (float)C_DIM) - mu * mu;
    const float rs  = rsqrtf(var + 1e-5f);

    _Float16* xo = xn + (size_t)row * C_DIM;
#pragma unroll
    for (int i = 0; i < 16; ++i) {
      const int c = lane + 32 * i;
      const float v = (vals[i] - mu) * rs * ln_w[c] + ln_b[c];
      xo[c] = (_Float16)v;
      T[ll * LNP + c] = v;
    }
  }
  __syncthreads();

  // Transposed, coalesced residual store: 512 channels x 32 rows = 4096 float4s.
  float* ro = resid + (size_t)bv * (L_SEQ * C_DIM) + l0;
  for (int g = (int)threadIdx.x; g < 4096; g += 256) {
    const int c   = g >> 3;
    const int llb = (g & 7) << 2;
    float4 v;
    v.x = T[(llb + 0) * LNP + c];
    v.y = T[(llb + 1) * LNP + c];
    v.z = T[(llb + 2) * LNP + c];
    v.w = T[(llb + 3) * LNP + c];
    *(float4*)(ro + (size_t)c * L_SEQ + llb) = v;
  }
}

// ---------------------------------------------------------------------------
// Kernel 2: f32 -> f16 weight conversion.
// ---------------------------------------------------------------------------
__global__ __launch_bounds__(256) void cvt_f16_kernel(const float* __restrict__ src,
                                                      _Float16* __restrict__ dst, int n) {
  const int i = blockIdx.x * blockDim.x + threadIdx.x;
  if (i < n) dst[i] = (_Float16)src[i];
}

// ---------------------------------------------------------------------------
// Kernel 3: projection GEMM with 32x32 register tiles per wave.
//   out[M,N] = A[M,512] * W[N,512]^T + bias
// 4 accumulators share 2 A-frags and 2 B-frags per k-step (4 WMMA / 4 loads).
// transposed==1 -> V^T store: out[bv, c, l] (f16).
// ---------------------------------------------------------------------------
__global__ __launch_bounds__(256) void proj_kernel(const _Float16* __restrict__ A,
                                                   const _Float16* __restrict__ W,
                                                   const float* __restrict__ bias,
                                                   _Float16* __restrict__ out,
                                                   int N, int transposed) {
  const int tiles_n = N >> 5;
  const int wave = blockIdx.x * (blockDim.x >> 5) + (threadIdx.x >> 5);
  const int tn = wave % tiles_n;
  const int tm = wave / tiles_n;
  const int m0 = tm << 5;
  const int n0 = tn << 5;

  const _Float16* Ab = A + (size_t)m0 * C_DIM;
  const _Float16* Bb = W + (size_t)n0 * C_DIM;

  v8f acc[2][2] = {};
#pragma unroll
  for (int k0 = 0; k0 < C_DIM; k0 += 32) {
    const v16h a0 = load_a_frag(Ab,              C_DIM, k0);
    const v16h a1 = load_a_frag(Ab + 16 * C_DIM, C_DIM, k0);
    const v16h b0 = load_b_frag(Bb,              C_DIM, k0);
    const v16h b1 = load_b_frag(Bb + 16 * C_DIM, C_DIM, k0);
    acc[0][0] = WMMA_F16(a0, b0, acc[0][0]);
    acc[0][1] = WMMA_F16(a0, b1, acc[0][1]);
    acc[1][0] = WMMA_F16(a1, b0, acc[1][0]);
    acc[1][1] = WMMA_F16(a1, b1, acc[1][1]);
  }

  const int lane = lane_id();
  const int n  = lane & 15;
  const int mo = (lane >> 4) << 3;     // row offset 0 or 8 (C/D layout)

#pragma unroll
  for (int mi = 0; mi < 2; ++mi) {
#pragma unroll
    for (int ni = 0; ni < 2; ++ni) {
      const int nn = n0 + 16 * ni + n;
      const float bn = bias[nn];
      if (!transposed) {
#pragma unroll
        for (int r = 0; r < 8; ++r)
          out[(size_t)(m0 + 16 * mi + mo + r) * N + nn] = (_Float16)(acc[mi][ni][r] + bn);
      } else {
        // V^T layout [BV][C][L]: (global row gm, channel c) -> [bv, c, l]
#pragma unroll
        for (int r = 0; r < 8; ++r) {
          const int gm = m0 + 16 * mi + mo + r;
          const int bv = gm >> 10;
          const int l  = gm & (L_SEQ - 1);
          out[((size_t)bv * C_DIM + nn) * L_SEQ + l] = (_Float16)(acc[mi][ni][r] + bn);
        }
      }
    }
  }
}

// ---------------------------------------------------------------------------
// Kernel 4: attention. One workgroup (8 waves) per (bv, 32-row Q block).
//  Phase 1: S[32,1024] = Qblk * K^T  (32x32 WMMA tiles; Q frags hoisted)
//  Phase 2: row max + exp; per-row 1/sum stashed in LDS (no normalize pass)
//  Phase 3: O = P * V; epilogue  out = gamma * invRow * acc + residual
// Dynamic LDS: 32*1024*4 + 32*4 bytes (~128 KB; WGP pool is 320 KB -> 2 WGs).
// ---------------------------------------------------------------------------
__global__ __launch_bounds__(256) void attn_kernel(const _Float16* __restrict__ Q,
                                                   const _Float16* __restrict__ K,
                                                   const _Float16* __restrict__ VT,
                                                   const float* __restrict__ gamma_p,
                                                   float* __restrict__ out) {
  extern __shared__ float S[];               // [32][L_SEQ] fp32 + [32] inv sums
  float* invA = S + 32 * L_SEQ;

  const int bv = blockIdx.x >> 5;            // 32 row blocks per bv
  const int m0 = (blockIdx.x & 31) << 5;
  const int wid  = (int)(threadIdx.x >> 5);
  const int lane = lane_id();

  const _Float16* Qb = Q + ((size_t)bv * L_SEQ + m0) * D_DIM;
  const _Float16* Kb = K + (size_t)bv * L_SEQ * D_DIM;

  const int n  = lane & 15;
  const int mo = (lane >> 4) << 3;

  // Hoist the 32x128 Q block into registers: 2 row-tiles x 4 k-steps.
  v16h qa[2][4];
#pragma unroll
  for (int mi = 0; mi < 2; ++mi)
#pragma unroll
    for (int kk = 0; kk < 4; ++kk)
      qa[mi][kk] = load_a_frag(Qb + (size_t)(16 * mi) * D_DIM, D_DIM, kk * 32);

  // ---- Phase 1: energy tiles -> LDS (32 col-tiles of 32, 4 per wave) ----
  for (int t = wid; t < (L_SEQ >> 5); t += 8) {
    const int j0 = t << 5;
    const _Float16* Bb = Kb + (size_t)j0 * D_DIM;
    __builtin_prefetch(Bb + ((size_t)256 * D_DIM), 0, 1);   // next-tile hint
    v8f acc[2][2] = {};
#pragma unroll
    for (int kk = 0; kk < 4; ++kk) {
      const v16h b0 = load_b_frag(Bb,              D_DIM, kk * 32);
      const v16h b1 = load_b_frag(Bb + 16 * D_DIM, D_DIM, kk * 32);
      acc[0][0] = WMMA_F16(qa[0][kk], b0, acc[0][0]);
      acc[0][1] = WMMA_F16(qa[0][kk], b1, acc[0][1]);
      acc[1][0] = WMMA_F16(qa[1][kk], b0, acc[1][0]);
      acc[1][1] = WMMA_F16(qa[1][kk], b1, acc[1][1]);
    }
#pragma unroll
    for (int mi = 0; mi < 2; ++mi)
#pragma unroll
      for (int ni = 0; ni < 2; ++ni)
#pragma unroll
        for (int r = 0; r < 8; ++r)
          S[(16 * mi + mo + r) * L_SEQ + j0 + 16 * ni + n] = acc[mi][ni][r];
  }
  __syncthreads();

  // ---- Phase 2: row max + exp; store 1/sum (8 waves x 4 rows) ----
#pragma unroll
  for (int rr = 0; rr < 4; ++rr) {
    const int row = wid * 4 + rr;
    float* Sr = &S[row * L_SEQ];
    float mx = -3.4e38f;
    for (int c = lane; c < L_SEQ; c += 32) mx = fmaxf(mx, Sr[c]);
#pragma unroll
    for (int off = 16; off; off >>= 1) mx = fmaxf(mx, __shfl_xor(mx, off, 32));
    float sum = 0.f;
    for (int c = lane; c < L_SEQ; c += 32) {
      const float e = __expf(Sr[c] - mx);
      Sr[c] = e;
      sum += e;
    }
#pragma unroll
    for (int off = 16; off; off >>= 1) sum += __shfl_xor(sum, off, 32);
    if (lane == 0) invA[row] = 1.0f / sum;
  }
  __syncthreads();

  // Per-wave row scales (gamma * 1/sum), hoisted out of the tile loop.
  const float g = gamma_p[0];
  float giv[2][8];
#pragma unroll
  for (int mi = 0; mi < 2; ++mi)
#pragma unroll
    for (int r = 0; r < 8; ++r)
      giv[mi][r] = g * invA[16 * mi + mo + r];

  // ---- Phase 3: O = P * V, 32x32 tiles (16 col-tiles, 2 per wave) ----
  float* Ob = out + ((size_t)bv * L_SEQ + m0) * C_DIM;
  for (int t = wid; t < (C_DIM >> 5); t += 8) {
    const int c0 = t << 5;
    const _Float16* Bb = VT + ((size_t)bv * C_DIM + c0) * L_SEQ;
    v8f acc[2][2] = {};
    for (int k0 = 0; k0 < L_SEQ; k0 += 32) {
      const v16h a0 = load_a_frag_f32(S,              L_SEQ, k0);  // LDS, cvt->f16
      const v16h a1 = load_a_frag_f32(S + 16 * L_SEQ, L_SEQ, k0);
      const v16h b0 = load_b_frag(Bb,                 L_SEQ, k0);
      const v16h b1 = load_b_frag(Bb + 16 * L_SEQ,    L_SEQ, k0);
      acc[0][0] = WMMA_F16(a0, b0, acc[0][0]);
      acc[0][1] = WMMA_F16(a0, b1, acc[0][1]);
      acc[1][0] = WMMA_F16(a1, b0, acc[1][0]);
      acc[1][1] = WMMA_F16(a1, b1, acc[1][1]);
    }
#pragma unroll
    for (int mi = 0; mi < 2; ++mi)
#pragma unroll
      for (int ni = 0; ni < 2; ++ni)
#pragma unroll
        for (int r = 0; r < 8; ++r) {
          float* p = Ob + (size_t)(16 * mi + mo + r) * C_DIM + c0 + 16 * ni + n;
          *p = giv[mi][r] * acc[mi][ni][r] + *p;   // residual already in d_out
        }
  }
}

// ---------------------------------------------------------------------------
// Host launcher.
// Inputs: x, ln_w, ln_b, wq, bq, wk, bk, wv, bv, gamma  (all fp32)
// Workspace layout (f16): XN 32MB | Q 8MB | K 8MB | VT 32MB | WQ16 | WK16 | WV16
// ---------------------------------------------------------------------------
extern "C" void kernel_launch(void* const* d_in, const int* in_sizes, int n_in,
                              void* d_out, int out_size, void* d_ws, size_t ws_size,
                              hipStream_t stream) {
  const float* x    = (const float*)d_in[0];
  const float* ln_w = (const float*)d_in[1];
  const float* ln_b = (const float*)d_in[2];
  const float* wq   = (const float*)d_in[3];
  const float* bq   = (const float*)d_in[4];
  const float* wk   = (const float*)d_in[5];
  const float* bk   = (const float*)d_in[6];
  const float* wv   = (const float*)d_in[7];
  const float* bvp  = (const float*)d_in[8];
  const float* gm   = (const float*)d_in[9];
  float* out = (float*)d_out;

  char* ws = (char*)d_ws;
  size_t off = 0;
  _Float16* XN = (_Float16*)(ws + off); off += (size_t)ROWS * C_DIM * 2;         // 32 MB
  _Float16* Qh = (_Float16*)(ws + off); off += (size_t)ROWS * D_DIM * 2;         //  8 MB
  _Float16* Kh = (_Float16*)(ws + off); off += (size_t)ROWS * D_DIM * 2;         //  8 MB
  _Float16* VT = (_Float16*)(ws + off); off += (size_t)BV_N * C_DIM * L_SEQ * 2; // 32 MB
  _Float16* WQ16 = (_Float16*)(ws + off); off += (size_t)D_DIM * C_DIM * 2;
  _Float16* WK16 = (_Float16*)(ws + off); off += (size_t)D_DIM * C_DIM * 2;
  _Float16* WV16 = (_Float16*)(ws + off); off += (size_t)C_DIM * C_DIM * 2;

  // 1) LayerNorm + coalesced transposed residual (32 rows per block)
  ln_kernel<<<ROWS / 32, 256, (size_t)32 * LNP * sizeof(float), stream>>>(
      x, ln_w, ln_b, XN, out);

  // 2) weight conversion
  cvt_f16_kernel<<<(D_DIM * C_DIM + 255) / 256, 256, 0, stream>>>(wq, WQ16, D_DIM * C_DIM);
  cvt_f16_kernel<<<(D_DIM * C_DIM + 255) / 256, 256, 0, stream>>>(wk, WK16, D_DIM * C_DIM);
  cvt_f16_kernel<<<(C_DIM * C_DIM + 255) / 256, 256, 0, stream>>>(wv, WV16, C_DIM * C_DIM);

  // 3) projections: one wave per 32x32 tile, 8 waves/block
  {
    const int tiles_q = (ROWS / 32) * (D_DIM / 32);   // 4096 waves
    proj_kernel<<<tiles_q / 8, 256, 0, stream>>>(XN, WQ16, bq, Qh, D_DIM, 0);
    proj_kernel<<<tiles_q / 8, 256, 0, stream>>>(XN, WK16, bk, Kh, D_DIM, 0);
    const int tiles_v = (ROWS / 32) * (C_DIM / 32);   // 16384 waves
    proj_kernel<<<tiles_v / 8, 256, 0, stream>>>(XN, WV16, bvp, VT, C_DIM, 1);
  }

  // 4) attention: one WG per (bv, 32-row block) = 1024 WGs, ~128 KB dynamic LDS
  attn_kernel<<<BV_N * (L_SEQ / 32), 256,
                ((size_t)32 * L_SEQ + 32) * sizeof(float), stream>>>(
      Qh, Kh, VT, gm, out);
}
